// MultiHeadDotProductAttention_39651138077436
// MI455X (gfx1250) — compile-verified, matrix operand order checked by reading, not run on hardware
//
#include <hip/hip_runtime.h>
#include <math.h>

// ---------------------------------------------------------------------------
// Types matching the gfx1250 WMMA builtin signatures
// ---------------------------------------------------------------------------
typedef __bf16 bf16_t;
typedef __attribute__((ext_vector_type(16))) __bf16 v16bf;
typedef __attribute__((ext_vector_type(8)))  float  v8f;

// ---------------------------------------------------------------------------
// Load a 16x32 bf16 WMMA fragment from a row-major tile in LDS.
// Per CDNA5 ISA 7.12.2 (16-bit A-matrix 16x32):
//   lanes 0-15 : row = lane,     K = {0..7} U {16..23}
//   lanes 16-31: row = lane-16,  K = {8..15} U {24..31}
// The B operand (Kx16) uses the same per-lane layout with lane = column,
// so we feed it B^T (i.e. Bt[N][K]) through the same loader.
// Each 8-element half is contiguous -> compiler emits ds_load_b128 pairs.
// ---------------------------------------------------------------------------
__device__ __forceinline__ v16bf frag_load(const bf16_t* __restrict__ base, int ld) {
  const int lane = threadIdx.x & 31;
  const int row  = lane & 15;
  const int kh   = (lane >> 4) << 3;  // 0 or 8
  const bf16_t* p = base + row * ld;
  v16bf f;
#pragma unroll
  for (int i = 0; i < 8; ++i) f[i] = p[kh + i];
#pragma unroll
  for (int i = 0; i < 8; ++i) f[8 + i] = p[16 + kh + i];
  return f;
}

__device__ __forceinline__ v8f wmma_bf16(v16bf a, v16bf b, v8f c) {
  return __builtin_amdgcn_wmma_f32_16x16x32_bf16(false, a, false, b,
                                                 (short)0, c, false, false);
}

// ---------------------------------------------------------------------------
// Elementwise f32 -> bf16 cast
// ---------------------------------------------------------------------------
__global__ void cast_bf16_kernel(const float* __restrict__ in,
                                 bf16_t* __restrict__ out, long long n) {
  long long i = (long long)blockIdx.x * blockDim.x + threadIdx.x;
  if (i < n) out[i] = (bf16_t)in[i];
}

// ---------------------------------------------------------------------------
// Batched transpose + cast: in[z][R][C] f32  ->  out[z][C][R] bf16
// blockIdx.y = z
// ---------------------------------------------------------------------------
__global__ void transpose_bf16_kernel(const float* __restrict__ in,
                                      bf16_t* __restrict__ out,
                                      int R, int C,
                                      long long inBatch, long long outBatch) {
  const int z = blockIdx.y;
  long long i = (long long)blockIdx.x * blockDim.x + threadIdx.x;
  long long n = (long long)R * C;
  if (i < n) {
    int c = (int)(i / R);
    int r = (int)(i % R);
    out[z * outBatch + i] = (bf16_t)in[z * inBatch + (long long)r * C + c];
  }
}

// ---------------------------------------------------------------------------
// Fold w_a (and 1/sqrt(D)) into w_q, producing transposed bf16 weight:
//   WqaT[h*64+j][e] = (1/8) * sum_i w_q[e][h*64+i] * w_a[h][i][j]
// ---------------------------------------------------------------------------
__global__ void fold_wa_kernel(const float* __restrict__ w_q,
                               const float* __restrict__ w_a,
                               bf16_t* __restrict__ WqaT) {
  long long i = (long long)blockIdx.x * blockDim.x + threadIdx.x;
  if (i >= 1024LL * 1024LL) return;
  const int row = (int)(i >> 10);   // h*64 + j
  const int e   = (int)(i & 1023);
  const int h = row >> 6;
  const int j = row & 63;
  float s = 0.f;
#pragma unroll 8
  for (int ii = 0; ii < 64; ++ii)
    s += w_q[(long long)e * 1024 + h * 64 + ii] * w_a[((long long)h * 64 + ii) * 64 + j];
  WqaT[(long long)row * 1024 + e] = (bf16_t)(s * 0.125f);   // 1/sqrt(64)
}

// ---------------------------------------------------------------------------
// Generic tiled bf16 WMMA GEMM:  C[z] = act( A[z] @ Bt[z]^T + bias[z] )
//   A  : [M][K] bf16 row-major       Bt : [N][K] bf16 row-major
//   Block tile 128x64, 8 waves (4x2); each wave owns a 32x32 output:
//     2 A frags + 2 B frags -> 4 WMMA per K-step (2x the FLOPs per LDS byte
//     and per barrier vs a 16x16 wave tile).
//   K must be a multiple of 32 (true for every call site here).
//   Next K-tile is prefetched with global_prefetch_b8 while WMMAs run.
//   flags: bit0 = SiLU epilogue, bit1 = f32 output (else bf16)
// ---------------------------------------------------------------------------
#define BM 128
#define BN 64
#define BK 32

__global__ void __launch_bounds__(256)
gemm_wmma_kernel(const bf16_t* __restrict__ A, const bf16_t* __restrict__ Bt,
                 void* __restrict__ Cv, const float* __restrict__ bias,
                 int M, int N, int K, int ldc,
                 long long aBatch, long long bBatch, long long cBatch,
                 int biasBatch, int flags) {
  __shared__ bf16_t sA[BM * BK];   // 8 KB
  __shared__ bf16_t sB[BN * BK];   // 4 KB
  const int tid = threadIdx.x;
  const int z = blockIdx.z;
  A  += (long long)z * aBatch;
  Bt += (long long)z * bBatch;
  const int m0 = blockIdx.x * BM;
  const int n0 = blockIdx.y * BN;
  const int wid = tid >> 5;
  const int wm = wid & 3;   // 0..3 -> 32-row strip
  const int wn = wid >> 2;  // 0..1 -> 32-col strip
  v8f acc00 = {}, acc01 = {}, acc10 = {}, acc11 = {};

  // Staging map: thread covers two 8-element chunks of A, one of B.
  const int liA0 = tid * 8;            // A: 4096 elems = 2 chunks/thread
  const int liA1 = tid * 8 + 2048;
  const int liB  = tid * 8;            // B: 2048 elems = 1 chunk/thread
  const int rA0 = liA0 >> 5, cA0 = liA0 & 31;
  const int rA1 = liA1 >> 5, cA1 = liA1 & 31;
  const int rB  = liB  >> 5, cB  = liB  & 31;

  for (int kb = 0; kb < K; kb += BK) {
    { // stage A 128x32
      const int gr0 = m0 + rA0;
      bf16_t* dst = sA + liA0;
      if (gr0 < M) {
        const bf16_t* src = A + (long long)gr0 * K + (kb + cA0);
#pragma unroll
        for (int j = 0; j < 8; ++j) dst[j] = src[j];
      } else {
#pragma unroll
        for (int j = 0; j < 8; ++j) dst[j] = (bf16_t)0.0f;
      }
      const int gr1 = m0 + rA1;
      dst = sA + liA1;
      if (gr1 < M) {
        const bf16_t* src = A + (long long)gr1 * K + (kb + cA1);
#pragma unroll
        for (int j = 0; j < 8; ++j) dst[j] = src[j];
      } else {
#pragma unroll
        for (int j = 0; j < 8; ++j) dst[j] = (bf16_t)0.0f;
      }
    }
    { // stage Bt 64x32
      const int gn = n0 + rB;
      bf16_t* dst = sB + liB;
      if (gn < N) {
        const bf16_t* src = Bt + (long long)gn * K + (kb + cB);
#pragma unroll
        for (int j = 0; j < 8; ++j) dst[j] = src[j];
      } else {
#pragma unroll
        for (int j = 0; j < 8; ++j) dst[j] = (bf16_t)0.0f;
      }
    }
    // Prefetch next K-tile into the cache hierarchy while we compute
    // (lowers to global_prefetch_b8; speculative, dropped if invalid).
    const int kn = kb + BK;
    if (kn < K) {
      if (m0 + rA0 < M) __builtin_prefetch(A + (long long)(m0 + rA0) * K + (kn + cA0), 0, 3);
      if (m0 + rA1 < M) __builtin_prefetch(A + (long long)(m0 + rA1) * K + (kn + cA1), 0, 3);
      if (n0 + rB  < N) __builtin_prefetch(Bt + (long long)(n0 + rB) * K + (kn + cB), 0, 3);
    }
    __syncthreads();
    v16bf a0 = frag_load(sA + (wm * 32) * BK, BK);
    v16bf a1 = frag_load(sA + (wm * 32 + 16) * BK, BK);
    v16bf b0 = frag_load(sB + (wn * 32) * BK, BK);
    v16bf b1 = frag_load(sB + (wn * 32 + 16) * BK, BK);
    acc00 = wmma_bf16(a0, b0, acc00);
    acc01 = wmma_bf16(a0, b1, acc01);
    acc10 = wmma_bf16(a1, b0, acc10);
    acc11 = wmma_bf16(a1, b1, acc11);
    __syncthreads();
  }

  // Epilogue.  C/D layout: VGPR r -> row (+ hi*8), lane&15 -> col.
  const int lane = tid & 31;
  const int ln = lane & 15;
  const int hi = lane >> 4;
  float*  Cf = (float*)Cv;
  bf16_t* Cb = (bf16_t*)Cv;
  const bool do_silu = (flags & 1) != 0;
  const bool out_f32 = (flags & 2) != 0;
#pragma unroll
  for (int si = 0; si < 2; ++si) {
#pragma unroll
    for (int sj = 0; sj < 2; ++sj) {
      v8f acc = si ? (sj ? acc11 : acc10) : (sj ? acc01 : acc00);
      const int n = n0 + wn * 32 + sj * 16 + ln;
#pragma unroll
      for (int r = 0; r < 8; ++r) {
        const int m = m0 + wm * 32 + si * 16 + hi * 8 + r;
        if (m < M && n < N) {
          float v = acc[r];
          if (bias) v += bias[(long long)biasBatch * z + n];
          if (do_silu) v = v * (1.0f / (1.0f + __expf(-v)));
          long long ci = (long long)cBatch * z + (long long)m * ldc + n;
          if (out_f32) Cf[ci] = v; else Cb[ci] = (bf16_t)v;
        }
      }
    }
  }
}

// ---------------------------------------------------------------------------
// Fused attention per (b,h):  scores = QA@K^T (+supp), softmax, ctx = P@V
// One 256-thread block per (b,h); S=79 padded to 80 rows / 96 K for WMMA.
// LDS overlay: sP reuses the sQ/sK region after the score phase.
// ---------------------------------------------------------------------------
__global__ void __launch_bounds__(256)
attn_wmma_kernel(const bf16_t* __restrict__ QA, const bf16_t* __restrict__ Km,
                 const bf16_t* __restrict__ Vm, const bf16_t* __restrict__ Supp,
                 bf16_t* __restrict__ Ctx) {
  constexpr int S = 79, SP = 80, SK = 96, D = 64, HD = 1024;
  __shared__ __align__(16) char smem[58368];
  bf16_t* sQ  = (bf16_t*)smem;                 // [80][64] 10240 B
  bf16_t* sKt = (bf16_t*)(smem + 10240);       // [80][64] 10240 B
  bf16_t* sVt = (bf16_t*)(smem + 20480);       // [64][96] 12288 B
  float*  sS  = (float*)(smem + 32768);        // [80][80] 25600 B
  bf16_t* sP  = (bf16_t*)smem;                 // [80][96] overlays sQ/sKt

  const int tid = threadIdx.x;
  const int bidx = blockIdx.x;       // b*16 + h
  const int b = bidx >> 4;
  const int h = bidx & 15;
  const long long rowBase = (long long)b * S * HD + (long long)h * D;

  // Stage QA and K tiles ([t][d], zero-padded rows >= 79).
  for (int i = tid; i < SP * D; i += 256) {
    const int t = i >> 6, c = i & 63;
    const bool ok = t < S;
    sQ[i]  = ok ? QA[rowBase + (long long)t * HD + c] : (bf16_t)0.0f;
    sKt[i] = ok ? Km[rowBase + (long long)t * HD + c] : (bf16_t)0.0f;
  }
  // Stage V transposed: sVt[d][t] so it serves as Bt for P@V.
  for (int i = tid; i < D * SK; i += 256) {
    const int d = i / SK, t = i % SK;
    sVt[i] = (t < S) ? Vm[rowBase + (long long)t * HD + d] : (bf16_t)0.0f;
  }
  __syncthreads();

  const int wid = tid >> 5;
  const int lane = tid & 31, ln = lane & 15, hi = lane >> 4;

  // Scores: QA[80x64] @ K^T -> sS[80][80]; 5x5 subtiles over 8 waves.
  for (int st = wid; st < 25; st += 8) {
    const int tm = st / 5, tn = st % 5;
    v8f acc = {};
#pragma unroll
    for (int k0 = 0; k0 < D; k0 += 32) {
      v16bf a  = frag_load(sQ  + (tm * 16) * D + k0, D);
      v16bf bb = frag_load(sKt + (tn * 16) * D + k0, D);   // K rows == Bt
      acc = wmma_bf16(a, bb, acc);
    }
#pragma unroll
    for (int r = 0; r < 8; ++r)
      sS[(tm * 16 + hi * 8 + r) * SP + (tn * 16 + ln)] = acc[r];
  }
  __syncthreads();

  // Softmax per row t (add smolgen supp logits); write P bf16 with zero pad.
  const bf16_t* supp = Supp + (long long)bidx * (S * S);
  for (int t = tid; t < SP; t += 256) {
    if (t < S) {
      float mx = -1e30f;
      for (int T = 0; T < S; ++T) {
        float v = sS[t * SP + T] + (float)supp[t * S + T];
        sS[t * SP + T] = v;
        mx = fmaxf(mx, v);
      }
      float sum = 0.f;
      for (int T = 0; T < S; ++T) {
        float e = __expf(sS[t * SP + T] - mx);
        sS[t * SP + T] = e;
        sum += e;
      }
      const float inv = 1.0f / sum;
      for (int T = 0; T < SK; ++T)
        sP[t * SK + T] = (T < S) ? (bf16_t)(sS[t * SP + T] * inv) : (bf16_t)0.0f;
    } else {
      for (int T = 0; T < SK; ++T) sP[t * SK + T] = (bf16_t)0.0f;
    }
  }
  __syncthreads();

  // ctx = P[80x96] @ V[96x64]; 5x4 subtiles over 8 waves; V^T staged as Bt.
  for (int st = wid; st < 20; st += 8) {
    const int tm = st >> 2, tn = st & 3;
    v8f acc = {};
#pragma unroll
    for (int k0 = 0; k0 < SK; k0 += 32) {
      v16bf a  = frag_load(sP  + (tm * 16) * SK + k0, SK);
      v16bf bb = frag_load(sVt + (tn * 16) * SK + k0, SK);
      acc = wmma_bf16(a, bb, acc);
    }
#pragma unroll
    for (int r = 0; r < 8; ++r) {
      const int t = tm * 16 + hi * 8 + r;
      const int d = tn * 16 + ln;
      if (t < S) Ctx[rowBase + (long long)t * HD + d] = (bf16_t)acc[r];
    }
  }
}

// ---------------------------------------------------------------------------
// Host-side orchestration
// ---------------------------------------------------------------------------
static inline int ceil_div(long long a, long long b) { return (int)((a + b - 1) / b); }

extern "C" void kernel_launch(void* const* d_in, const int* in_sizes, int n_in,
                              void* d_out, int out_size, void* d_ws, size_t ws_size,
                              hipStream_t stream) {
  (void)in_sizes; (void)n_in; (void)out_size; (void)ws_size;
  const float* in_q    = (const float*)d_in[0];
  const float* in_kv   = (const float*)d_in[1];
  const float* w_q     = (const float*)d_in[2];
  const float* w_k     = (const float*)d_in[3];
  const float* w_v     = (const float*)d_in[4];
  const float* w_o     = (const float*)d_in[5];
  const float* w_a     = (const float*)d_in[6];
  const float* w_comp  = (const float*)d_in[7];
  const float* w_pos   = (const float*)d_in[8];
  const float* b_pos   = (const float*)d_in[9];
  const float* w_head  = (const float*)d_in[10];
  const float* b_head  = (const float*)d_in[11];
  const float* s_proj  = (const float*)d_in[12];
  float* out = (float*)d_out;

  const long long R = 40448;      // B*S
  const int SS = 6241;            // S*S

  char* ws = (char*)d_ws;
  size_t off = 0;
  auto alloc = [&](long long elems) -> bf16_t* {
    bf16_t* p = (bf16_t*)(ws + off);
    off += (size_t)(((elems * 2 + 255) / 256) * 256);
    return p;
  };
  bf16_t* Xq     = alloc(R * 1024);
  bf16_t* Xkv    = alloc(R * 1024);
  bf16_t* WqaT   = alloc(1024LL * 1024);
  bf16_t* WkT    = alloc(1024LL * 1024);
  bf16_t* WvT    = alloc(1024LL * 1024);
  bf16_t* WoT    = alloc(1024LL * 1024);
  bf16_t* WcT    = alloc(32LL * 1024);
  bf16_t* WposT  = alloc(256LL * 2528);
  bf16_t* WheadT = alloc(16LL * 256 * 256);
  bf16_t* SprojT = alloc((long long)SS * 256);
  bf16_t* QA     = alloc(R * 1024);
  bf16_t* Km     = alloc(R * 1024);
  bf16_t* Vm     = alloc(R * 1024);
  bf16_t* comp   = alloc(R * 32);            // == [512][2528]
  bf16_t* posb   = alloc(512LL * 256);
  bf16_t* headb  = alloc(8192LL * 256);
  bf16_t* supp   = alloc(8192LL * (long long)SS);
  bf16_t* ctx    = alloc(R * 1024);

  // --- precision conversion + weight transposes -----------------------------
  cast_bf16_kernel<<<ceil_div(R * 1024, 256), 256, 0, stream>>>(in_q,  Xq,  R * 1024);
  cast_bf16_kernel<<<ceil_div(R * 1024, 256), 256, 0, stream>>>(in_kv, Xkv, R * 1024);

  transpose_bf16_kernel<<<dim3(ceil_div(1024LL * 1024, 256), 1), 256, 0, stream>>>(
      w_k, WkT, 1024, 1024, 0, 0);
  transpose_bf16_kernel<<<dim3(ceil_div(1024LL * 1024, 256), 1), 256, 0, stream>>>(
      w_v, WvT, 1024, 1024, 0, 0);
  transpose_bf16_kernel<<<dim3(ceil_div(1024LL * 1024, 256), 1), 256, 0, stream>>>(
      w_o, WoT, 1024, 1024, 0, 0);
  transpose_bf16_kernel<<<dim3(ceil_div(1024LL * 32, 256), 1), 256, 0, stream>>>(
      w_comp, WcT, 1024, 32, 0, 0);
  transpose_bf16_kernel<<<dim3(ceil_div(2528LL * 256, 256), 1), 256, 0, stream>>>(
      w_pos, WposT, 2528, 256, 0, 0);
  transpose_bf16_kernel<<<dim3(ceil_div(256LL * 256, 256), 16), 256, 0, stream>>>(
      w_head, WheadT, 256, 256, 65536, 65536);
  transpose_bf16_kernel<<<dim3(ceil_div(256LL * SS, 256), 1), 256, 0, stream>>>(
      s_proj, SprojT, 256, SS, 0, 0);

  fold_wa_kernel<<<ceil_div(1024LL * 1024, 256), 256, 0, stream>>>(w_q, w_a, WqaT);

  // --- projections: QA (w_a folded, pre-scaled), K, V ------------------------
  gemm_wmma_kernel<<<dim3(316, 16, 1), 256, 0, stream>>>(
      Xq,  WqaT, QA, nullptr, (int)R, 1024, 1024, 1024, 0, 0, 0, 0, 0);
  gemm_wmma_kernel<<<dim3(316, 16, 1), 256, 0, stream>>>(
      Xkv, WkT,  Km, nullptr, (int)R, 1024, 1024, 1024, 0, 0, 0, 0, 0);
  gemm_wmma_kernel<<<dim3(316, 16, 1), 256, 0, stream>>>(
      Xkv, WvT,  Vm, nullptr, (int)R, 1024, 1024, 1024, 0, 0, 0, 0, 0);

  // --- smolgen path ----------------------------------------------------------
  // comp[b,s,c] = x @ w_compress   (contiguous == [512][2528])
  gemm_wmma_kernel<<<dim3(316, 1, 1), 256, 0, stream>>>(
      Xq, WcT, comp, nullptr, (int)R, 32, 1024, 32, 0, 0, 0, 0, 0);
  // pos = silu(comp @ w_pos + b_pos)   [512][256]
  gemm_wmma_kernel<<<dim3(4, 4, 1), 256, 0, stream>>>(
      comp, WposT, posb, b_pos, 512, 256, 2528, 256, 0, 0, 0, 0, 1);
  // head[b,h,:] = silu(pos @ w_head[h] + b_head[h])  -> rows (b*16+h)
  gemm_wmma_kernel<<<dim3(4, 4, 16), 256, 0, stream>>>(
      posb, WheadT, headb, b_head, 512, 256, 256, 4096,
      0, 65536, 256, 256, 1);
  // supp[(b*16+h)][t*79+T] = head @ shared_proj   [8192][6241]
  gemm_wmma_kernel<<<dim3(64, ceil_div(SS, 64), 1), 256, 0, stream>>>(
      headb, SprojT, supp, nullptr, 8192, SS, 256, SS, 0, 0, 0, 0, 0);

  // --- fused attention (scores + supp + softmax + P@V) -----------------------
  attn_wmma_kernel<<<8192, 256, 0, stream>>>(QA, Km, Vm, supp, ctx);

  // --- output projection (f32 out) -------------------------------------------
  gemm_wmma_kernel<<<dim3(316, 16, 1), 256, 0, stream>>>(
      ctx, WoT, out, nullptr, (int)R, 1024, 1024, 1024, 0, 0, 0, 0, 2);
}
